// VQ_VAE_58523224375329
// MI455X (gfx1250) — matrix operand order repaired and checked by reading, not compiled, not run on hardware
//
#include <hip/hip_runtime.h>
#include <hip/hip_bf16.h>

typedef __bf16 bf16_t;
typedef bf16_t v16bf __attribute__((ext_vector_type(16)));
typedef bf16_t v8bf  __attribute__((ext_vector_type(8)));
typedef float  v8f   __attribute__((ext_vector_type(8)));
typedef unsigned u32x4 __attribute__((ext_vector_type(4)));

__device__ __forceinline__ bf16_t f2bf(float x) {
    unsigned u = __builtin_bit_cast(unsigned, x);
    unsigned r = (u + 0x7FFFu + ((u >> 16) & 1u)) >> 16;   // RNE
    unsigned short s = (unsigned short)r;
    return __builtin_bit_cast(bf16_t, s);
}
__device__ __forceinline__ v8bf ld8(const bf16_t* p) {
    return *(const v8bf*)p;                                 // global_load_b128
}
__device__ __forceinline__ v8bf bz8() {
    u32x4 z = {0u, 0u, 0u, 0u};
    return __builtin_bit_cast(v8bf, z);
}
__device__ __forceinline__ v16bf cat16(v8bf a, v8bf b) {
    return __builtin_shufflevector(a, b, 0,1,2,3,4,5,6,7,8,9,10,11,12,13,14,15);
}

// ---------------------------------------------------------------------------
// Implicit-GEMM conv, all layouts pre-transformed:
//   act : NHWC bf16, channels = Cin (multiple of 32, zero padded)
//   wt  : [ocPad][KH*KW][Cin] bf16 (zero padded, spatial flip pre-applied)
// One wave owns a 64(oc) x 16(pixel) D-tile: MT=4 accumulators share one
// gathered B fragment per 32-wide K chunk. All fragment loads are contiguous
// b128 matching the CDNA5 16-bit A(16x32)/B(32x16) VGPR layouts.
// ---------------------------------------------------------------------------
template<int KH, int KW, int STRIDE, int PAD, int LDIL, int RELU, int MT>
__global__ __launch_bounds__(32)
void conv_wmma(const bf16_t* __restrict__ act, const bf16_t* __restrict__ wt,
               const float* __restrict__ bias,
               bf16_t* __restrict__ outb,   // NHWC bf16 (or null)
               float*  __restrict__ outf,   // NHWC fp32 (or null)
               int Cin, int Hin, int Win,
               int CoutReal, int Hout, int Wout)
{
    const int lane  = threadIdx.x & 31;
    const int half_ = lane >> 4;
    const int lrow  = lane & 15;
    constexpr int KHW = KH * KW;

    const int ocBase = blockIdx.x * (16 * MT);
    const int p0     = blockIdx.y * 16;
    const int HWo    = Hout * Wout;
    const int khwcin = KHW * Cin;

    // this lane's output pixel (B-fragment column N = lrow; D column = lrow)
    const int pn = p0 + lrow;
    const int pb = pn / HWo;
    const int ps = pn - pb * HWo;
    const int oh = ps / Wout;
    const int ow = ps - oh * Wout;

    // per-tap gather base + validity (constants fold via template params)
    int  bbase[KHW];
    bool bval[KHW];
    #pragma unroll
    for (int kh = 0; kh < KH; ++kh) {
        int  ihn = oh * STRIDE - PAD + kh;
        int  ih  = (LDIL == 1) ? ihn : (ihn / LDIL);
        bool vh  = (ihn >= 0) && ((LDIL == 1) || ((ihn % LDIL) == 0)) && (ih < Hin);
        #pragma unroll
        for (int kw = 0; kw < KW; ++kw) {
            int  iwn = ow * STRIDE - PAD + kw;
            int  iw  = (LDIL == 1) ? iwn : (iwn / LDIL);
            bool vw  = (iwn >= 0) && ((LDIL == 1) || ((iwn % LDIL) == 0)) && (iw < Win);
            bbase[kh * KW + kw] = ((pb * Hin + ih) * Win + iw) * Cin;
            bval [kh * KW + kw] = vh && vw;
        }
    }

    v8f acc[MT];
    #pragma unroll
    for (int m = 0; m < MT; ++m)
        #pragma unroll
        for (int r = 0; r < 8; ++r) acc[m][r] = 0.0f;

    #pragma unroll
    for (int t = 0; t < KHW; ++t) {
        const bf16_t* bp = act + bbase[t] + 8 * half_;
        const bool    ok = bval[t];
        for (int c0 = 0; c0 < Cin; c0 += 32) {
            // B fragment: runs of 8 at k = 8*half_ and 16+8*half_
            v8bf blo = ok ? ld8(bp + c0)     : bz8();
            v8bf bhi = ok ? ld8(bp + c0 + 16) : bz8();
            v16bf bf = cat16(blo, bhi);
            #pragma unroll
            for (int m = 0; m < MT; ++m) {
                // A fragment: runs of 8 at k = 8*half_ and 16+8*half_
                const bf16_t* wr = wt + (size_t)(ocBase + m * 16 + lrow) * khwcin
                                      + t * Cin + c0 + 8 * half_;
                v16bf af = cat16(ld8(wr), ld8(wr + 16));
                acc[m] = __builtin_amdgcn_wmma_f32_16x16x32_bf16(
                             false, af, false, bf, (short)0, acc[m], false, false);
            }
        }
    }

    // epilogue: D layout -> VGPR r = (M = r + 8*half_, N = lrow)
    #pragma unroll
    for (int m = 0; m < MT; ++m) {
        #pragma unroll
        for (int r = 0; r < 8; ++r) {
            int oc = ocBase + m * 16 + r + 8 * half_;
            if (oc < CoutReal) {
                float v = acc[m][r] + bias[oc];
                if (RELU) v = fmaxf(v, 0.0f);
                if (outb) outb[(size_t)pn * CoutReal + oc] = f2bf(v);
                if (outf) outf[(size_t)pn * CoutReal + oc] = v;
            }
        }
    }
}

// NCHW fp32 -> NHWC bf16 with channel zero-padding
__global__ __launch_bounds__(256)
void act_xform(const float* __restrict__ in, bf16_t* __restrict__ out,
               int B, int C, int HW, int Cpad)
{
    int e = blockIdx.x * blockDim.x + threadIdx.x;
    if (e >= B * HW * Cpad) return;
    int c   = e % Cpad;
    int pix = e / Cpad;
    int b   = pix / HW;
    int s   = pix - b * HW;
    out[e] = f2bf((c < C) ? in[((size_t)(b * C + c)) * HW + s] : 0.0f);
}

// OIHW fp32 -> [ocPad][KH*KW][Ipad] bf16, optional spatial flip, zero pad
__global__ __launch_bounds__(256)
void wt_xform(const float* __restrict__ w, bf16_t* __restrict__ out,
              int O, int I, int KH, int KW, int Opad, int Ipad, int flip)
{
    int KHW = KH * KW;
    int e = blockIdx.x * blockDim.x + threadIdx.x;
    if (e >= Opad * KHW * Ipad) return;
    int ci = e % Ipad;
    int r  = e / Ipad;
    int t  = r % KHW;
    int oc = r / KHW;
    int kh = t / KW, kw = t - kh * KW;
    if (flip) { kh = KH - 1 - kh; kw = KW - 1 - kw; }
    float v = (oc < O && ci < I) ? w[((size_t)(oc * I + ci) * KH + kh) * KW + kw] : 0.0f;
    out[e] = f2bf(v);
}

// codebook fp32 (K,64) -> bf16 copy + per-code squared norms
__global__ __launch_bounds__(256)
void cb_prep(const float* __restrict__ cb, bf16_t* __restrict__ cbbf,
             float* __restrict__ cnorm, int K)
{
    int k = blockIdx.x * blockDim.x + threadIdx.x;
    if (k >= K) return;
    const float* c = cb + (size_t)k * 64;
    float n = 0.0f;
    #pragma unroll 8
    for (int d = 0; d < 64; ++d) {
        float v = c[d];
        n = fmaf(v, v, n);
        cbbf[(size_t)k * 64 + d] = f2bf(v);
    }
    cnorm[k] = n;
}

// ---------------------------------------------------------------------------
// WMMA codebook search: d2 = ||z||^2 + ||c||^2 - 2 z.c, cross term via
// v_wmma (M=16 codes, N=16 pixels, K=64 as two K=32 chunks).
// One wave per 16-pixel tile; B(z) fragments loaded once, 32 code tiles.
// Lane argmin over its 8 code rows, shfl_xor(16) merges the half groups.
// ---------------------------------------------------------------------------
__global__ __launch_bounds__(32)
void vq_wmma(const bf16_t* __restrict__ zbf,  // (Npix,64) bf16
             const float*  __restrict__ z_e,  // (Npix,64) fp32 (exact ||z||^2)
             const bf16_t* __restrict__ cbbf, // (K,64) bf16
             const float*  __restrict__ cnorm,// (K)
             const float*  __restrict__ cb,   // (K,64) fp32 for z_q emit
             bf16_t* __restrict__ z_q,        // (Npix,64) bf16
             float* __restrict__ loss, int K, float scale)
{
    const int lane  = threadIdx.x & 31;
    const int half_ = lane >> 4;
    const int lrow  = lane & 15;
    const int pix   = blockIdx.x * 16 + lrow;

    // B fragments (z column = this lane's pixel): runs of 8 at k = k0+8*half_ and k0+16+8*half_
    const bf16_t* zp = zbf + (size_t)pix * 64 + 8 * half_;
    v16bf b0 = cat16(ld8(zp),      ld8(zp + 16));   // k chunk 0..31
    v16bf b1 = cat16(ld8(zp + 32), ld8(zp + 48));   // k chunk 32..63

    float best = 3.4e38f;
    int   bi   = 0;
    for (int ct = 0; ct < K / 16; ++ct) {
        const bf16_t* wr = cbbf + (size_t)(ct * 16 + lrow) * 64;
        v16bf a0 = cat16(ld8(wr +      8 * half_), ld8(wr + 16 + 8 * half_));
        v16bf a1 = cat16(ld8(wr + 32 + 8 * half_), ld8(wr + 48 + 8 * half_));
        v8f acc;
        #pragma unroll
        for (int r = 0; r < 8; ++r) acc[r] = 0.0f;
        acc = __builtin_amdgcn_wmma_f32_16x16x32_bf16(false, a0, false, b0,
                                                      (short)0, acc, false, false);
        acc = __builtin_amdgcn_wmma_f32_16x16x32_bf16(false, a1, false, b1,
                                                      (short)0, acc, false, false);
        #pragma unroll
        for (int r = 0; r < 8; ++r) {
            int code = ct * 16 + r + 8 * half_;
            float d  = cnorm[code] - 2.0f * acc[r];
            if (d < best) { best = d; bi = code; }
        }
    }
    // merge the two half-lane groups (same pixel column, disjoint code rows)
    float ob = __shfl_xor(best, 16, 32);
    int   oi = __shfl_xor(bi,   16, 32);
    if (ob < best) { best = ob; bi = oi; }

    if (lane < 16) {
        const float* zr = z_e + (size_t)pix * 64;
        float zn = 0.0f;
        #pragma unroll 8
        for (int d = 0; d < 64; ++d) zn = fmaf(zr[d], zr[d], zn);
        const float* cr = cb + (size_t)bi * 64;
        bf16_t* qp = z_q + (size_t)pix * 64;
        #pragma unroll 8
        for (int d = 0; d < 64; ++d) qp[d] = f2bf(cr[d]);
        atomicAdd(loss, scale * sqrtf(fmaxf(zn + best, 0.0f)));
    }
}

// Gaussian NLL on NHWC fp32 decoder output, wave32 shuffle tree + atomicAdd
__global__ __launch_bounds__(256)
void recon_kernel(const float* __restrict__ x, const float* __restrict__ o6,
                  float* __restrict__ loss, int Bn, int C, int HW2, float inv)
{
    const float LOG2PI_ = 0.9189385332046727f;
    int total = Bn * C * HW2;
    float local = 0.0f;
    for (int e = blockIdx.x * blockDim.x + threadIdx.x; e < total;
         e += gridDim.x * blockDim.x) {
        int b = e / (C * HW2);
        int r = e - b * (C * HW2);
        int c = r / HW2;
        int s = r - c * HW2;
        size_t pix = (size_t)b * HW2 + s;
        float mu = o6[pix * 6 + c];
        float lg = tanhf(o6[pix * 6 + 3 + c]);
        float df = x[e] - mu;
        local += df * df * 0.5f * __expf(-2.0f * lg) + LOG2PI_ + lg;
    }
    #pragma unroll
    for (int off = 16; off > 0; off >>= 1)
        local += __shfl_down(local, off, 32);
    if ((threadIdx.x & 31) == 0) atomicAdd(loss, local * inv);
}

__global__ void zero_kernel(float* p) {
    if (threadIdx.x == 0 && blockIdx.x == 0) p[0] = 0.0f;
}

// ---------------------------------------------------------------------------
extern "C" void kernel_launch(void* const* d_in, const int* in_sizes, int n_in,
                              void* d_out, int out_size, void* d_ws, size_t ws_size,
                              hipStream_t stream)
{
    const float* x   = (const float*)d_in[0];
    const float* ew1 = (const float*)d_in[1];
    const float* eb1 = (const float*)d_in[2];
    const float* ew2 = (const float*)d_in[3];
    const float* eb2 = (const float*)d_in[4];
    const float* ew3 = (const float*)d_in[5];
    const float* eb3 = (const float*)d_in[6];
    const float* cbk = (const float*)d_in[7];
    const float* dw1 = (const float*)d_in[8];
    const float* db1 = (const float*)d_in[9];
    const float* dw2 = (const float*)d_in[10];
    const float* db2 = (const float*)d_in[11];
    const float* dw3 = (const float*)d_in[12];
    const float* db3 = (const float*)d_in[13];
    float* loss = (float*)d_out;

    const int B = 64, C = 3, HID = 256, D = 64, K = 512;

    char* ws = (char*)d_ws;
    bf16_t* xb   = (bf16_t*)(ws);                       // 16 MB: (B,64,64,32) bf16
    bf16_t* h1   = (bf16_t*)(ws + (size_t)16777216);    // 32 MB: (B,32,32,256)
    bf16_t* h2   = (bf16_t*)(ws + (size_t)50331648);    //  8 MB: (B,16,16,256)
    float*  z_e  = (float*) (ws + (size_t)58720256);    //  4 MB: (B,16,16,64) fp32
    bf16_t* z_q  = (bf16_t*)(ws + (size_t)62914560);    //  2 MB: (B,16,16,64)
    float*  o6   = (float*) (ws + (size_t)65011712);    //  6 MB: (B,64,64,6) fp32
    bf16_t* w1   = (bf16_t*)(ws + (size_t)71303168);    // 256*16*32
    bf16_t* w2   = (bf16_t*)(ws + (size_t)71565312);    // 256*16*256
    bf16_t* w3   = (bf16_t*)(ws + (size_t)73662464);    // 64*9*256
    bf16_t* wd1  = (bf16_t*)(ws + (size_t)73957376);    // 256*9*64
    bf16_t* wd2  = (bf16_t*)(ws + (size_t)74252288);    // 256*16*256
    bf16_t* wd3  = (bf16_t*)(ws + (size_t)76349440);    // 64*16*256
    bf16_t* zbf  = (bf16_t*)(ws + (size_t)76873728);    //  2 MB: (B,16,16,64) bf16
    bf16_t* cbbf = (bf16_t*)(ws + (size_t)78970880);    // 64 KB: (512,64) bf16
    float*  cnrm = (float*) (ws + (size_t)79036416);    //  2 KB: (512)
    bf16_t* g1   = h2;                                  // reuse (h2 dead)
    bf16_t* g2   = h1;                                  // reuse (h1 dead)

    zero_kernel<<<1, 32, 0, stream>>>(loss);

    // ---- one-pass layout transforms (pure bandwidth, ~5us at 23.3 TB/s) ----
    act_xform<<<(B*64*64*32)/256, 256, 0, stream>>>(x, xb, B, C, 64*64, 32);
    wt_xform<<<(256*16*32  + 255)/256, 256, 0, stream>>>(ew1, w1,  HID, C,   4,4, HID, 32,  0);
    wt_xform<<<(256*16*256 + 255)/256, 256, 0, stream>>>(ew2, w2,  HID, HID, 4,4, HID, HID, 0);
    wt_xform<<<(64*9*256   + 255)/256, 256, 0, stream>>>(ew3, w3,  D,   HID, 3,3, D,   HID, 0);
    wt_xform<<<(256*9*64   + 255)/256, 256, 0, stream>>>(dw1, wd1, HID, D,   3,3, HID, D,   0);
    wt_xform<<<(256*16*256 + 255)/256, 256, 0, stream>>>(dw2, wd2, HID, HID, 4,4, HID, HID, 1);
    wt_xform<<<(64*16*256  + 255)/256, 256, 0, stream>>>(dw3, wd3, 2*C, HID, 4,4, 64,  HID, 1);
    cb_prep<<<(K + 255)/256, 256, 0, stream>>>(cbk, cbbf, cnrm, K);

    dim3 blk(32, 1, 1);
    // ---- encoder ----
    conv_wmma<4,4,2,1,1,1,4><<<dim3(4, (B*32*32)/16), blk, 0, stream>>>(
        xb, w1, eb1, h1, nullptr, 32, 64, 64, HID, 32, 32);
    conv_wmma<4,4,2,1,1,1,4><<<dim3(4, (B*16*16)/16), blk, 0, stream>>>(
        h1, w2, eb2, h2, nullptr, HID, 32, 32, HID, 16, 16);
    conv_wmma<3,3,1,1,1,0,4><<<dim3(1, (B*16*16)/16), blk, 0, stream>>>(
        h2, w3, eb3, zbf, z_e, HID, 16, 16, D, 16, 16);

    // ---- vector quantization (WMMA distance GEMM + argmin) ----
    vq_wmma<<<(B*16*16)/16, blk, 0, stream>>>(
        zbf, z_e, cbbf, cnrm, cbk, z_q, loss, K, 1.25f / (float)B);

    // ---- decoder (tconv = input-dilated conv, flip folded into weights) ----
    conv_wmma<3,3,1,1,1,1,4><<<dim3(4, (B*16*16)/16), blk, 0, stream>>>(
        z_q, wd1, db1, g1, nullptr, D, 16, 16, HID, 16, 16);
    conv_wmma<4,4,1,2,2,1,4><<<dim3(4, (B*32*32)/16), blk, 0, stream>>>(
        g1, wd2, db2, g2, nullptr, HID, 16, 16, HID, 32, 32);
    conv_wmma<4,4,1,2,2,0,4><<<dim3(1, (B*64*64)/16), blk, 0, stream>>>(
        g2, wd3, db3, nullptr, o6, HID, 32, 32, 2*C, 64, 64);

    // ---- reconstruction NLL ----
    recon_kernel<<<512, 256, 0, stream>>>(
        x, o6, loss, B, C, 64*64, 1.0f / (float)(B * C * 64 * 64));
}
